// SCMF_12008728559880
// MI455X (gfx1250) — 16-bit and 8-bit matrix operand order repaired and checked by reading, compiled, not run on hardware
//
#include <hip/hip_runtime.h>
#include <hip/hip_bf16.h>

#define NB   4
#define CCH  256
#define HH   96
#define WW   96
#define HW   (HH*WW)
#define NHEADS 8
#define CHD  32
#define DD   3072

typedef __attribute__((ext_vector_type(2)))  float     v2f_t;
typedef __attribute__((ext_vector_type(8)))  int       v8i_t;
typedef __attribute__((ext_vector_type(8)))  float     v8f_t;
typedef __attribute__((ext_vector_type(16))) _Float16  v16h_t;

#if defined(__has_builtin)
#  if __has_builtin(__builtin_amdgcn_wmma_f32_16x16x4_f32)
#    define HAVE_WMMA_F32 1
#  else
#    define HAVE_WMMA_F32 0
#  endif
#else
#  define HAVE_WMMA_F32 0
#endif

__device__ __forceinline__ float wred_sum(float v) {
#pragma unroll
  for (int o = 16; o > 0; o >>= 1) v += __shfl_xor(v, o, 32);
  return v;
}
__device__ __forceinline__ float wred_max(float v) {
#pragma unroll
  for (int o = 16; o > 0; o >>= 1) v = fmaxf(v, __shfl_xor(v, o, 32));
  return v;
}

__global__ __launch_bounds__(256)
void layernorm_k(const float* __restrict__ x, const float* __restrict__ w,
                 const float* __restrict__ bb, float* __restrict__ y) {
  __shared__ float sh[16];
  const int pos = blockIdx.x;
  const int b = pos / HW, s = pos % HW;
  const int c = threadIdx.x;
  const float v = x[((size_t)b * CCH + c) * HW + s];
  float s1 = wred_sum(v), s2 = wred_sum(v * v);
  if ((threadIdx.x & 31) == 0) { sh[threadIdx.x >> 5] = s1; sh[8 + (threadIdx.x >> 5)] = s2; }
  __syncthreads();
  float mu = 0.f, m2 = 0.f;
#pragma unroll
  for (int i = 0; i < 8; ++i) { mu += sh[i]; m2 += sh[8 + i]; }
  mu *= (1.0f / CCH);
  m2 = m2 * (1.0f / CCH) - mu * mu;
  const float inv = rsqrtf(m2 + 1e-5f);
  y[((size_t)b * CCH + c) * HW + s] = (v - mu) * inv * w[c] + bb[c];
}

__global__ __launch_bounds__(256)
void gap_k(const float* __restrict__ x, float* __restrict__ g) {
  __shared__ float sh[8];
  const int bc = blockIdx.x;
  const float* p = x + (size_t)bc * HW;
  float s = 0.f;
  for (int i = threadIdx.x; i < HW; i += 256) s += p[i];
  s = wred_sum(s);
  if ((threadIdx.x & 31) == 0) sh[threadIdx.x >> 5] = s;
  __syncthreads();
  if (threadIdx.x == 0) {
    float t = 0.f;
#pragma unroll
    for (int i = 0; i < 8; ++i) t += sh[i];
    g[bc] = t * (1.0f / HW);
  }
}

__global__ __launch_bounds__(256)
void se_fc_k(const float* __restrict__ gap, const float* __restrict__ fc1,
             const float* __restrict__ fc2, float* __restrict__ att) {
  __shared__ float y1[32];
  const int b = blockIdx.x;
  if (threadIdx.x < 32) {
    const int r = threadIdx.x;
    float s = 0.f;
    for (int c = 0; c < CCH; ++c) s += gap[b * CCH + c] * fc1[r * CCH + c];
    y1[r] = fmaxf(s, 0.f);
  }
  __syncthreads();
  const int c = threadIdx.x;
  float s = 0.f;
#pragma unroll
  for (int r = 0; r < 32; ++r) s += y1[r] * fc2[c * 32 + r];
  att[b * CCH + c] = 1.0f / (1.0f + __expf(-s));
}

__global__ __launch_bounds__(256)
void oute_k(const float* __restrict__ xe, const float* __restrict__ att,
            const float* __restrict__ pw, const float* __restrict__ pb,
            float* __restrict__ oe) {
  const int idx = blockIdx.x * 256 + threadIdx.x;
  if (idx >= NB * CCH * HW) return;
  const int b = idx / (CCH * HW);
  const int c = (idx / HW) % CCH;
  oe[idx] = xe[idx] * att[b * CCH + c] * pw[c] + pb[c];
}

__global__ __launch_bounds__(256)
void outr_k(const float* __restrict__ xr,
            const float* __restrict__ r1w, const float* __restrict__ r1b,
            const float* __restrict__ r2w, const float* __restrict__ r2b,
            const float* __restrict__ r3w, const float* __restrict__ r3b,
            const float* __restrict__ r4w, const float* __restrict__ r4b,
            const float* __restrict__ r5w, const float* __restrict__ r5b,
            const float* __restrict__ r6w, const float* __restrict__ r6b,
            const float* __restrict__ pw, const float* __restrict__ pb,
            float* __restrict__ orr) {
  const int idx = blockIdx.x * 256 + threadIdx.x;
  if (idx >= NB * CCH * HW) return;
  const int b = idx / (CCH * HW);
  const int c = (idx / HW) % CCH;
  const int h = (idx / WW) % HH;
  const int w = idx % WW;
  const float* base = xr + ((size_t)b * CCH + c) * HW;
  float acc = r1b[c] + r2b[c] + r3b[c] + r4b[c] + r5b[c] + r6b[c];
#pragma unroll
  for (int d = -3; d <= 3; ++d) {
    const int ww2 = w + d;
    if ((unsigned)ww2 < (unsigned)WW) {
      float wt = r5w[c * 7 + d + 3];
      if (d >= -2 && d <= 2) wt += r3w[c * 5 + d + 2];
      if (d >= -1 && d <= 1) wt += r1w[c * 3 + d + 1];
      acc += base[h * WW + ww2] * wt;
    }
  }
#pragma unroll
  for (int d = -3; d <= 3; ++d) {
    const int hh2 = h + d;
    if ((unsigned)hh2 < (unsigned)HH) {
      float wt = r6w[c * 7 + d + 3];
      if (d >= -2 && d <= 2) wt += r4w[c * 5 + d + 2];
      if (d >= -1 && d <= 1) wt += r2w[c * 3 + d + 1];
      acc += base[hh2 * WW + w] * wt;
    }
  }
  orr[idx] = acc * pw[c] + pb[c];
}

__global__ __launch_bounds__(256)
void spike_rearrange_k(const float* __restrict__ oe, const float* __restrict__ orr,
                       unsigned char* __restrict__ Ehw, unsigned char* __restrict__ Rhw,
                       unsigned char* __restrict__ Ewh, unsigned char* __restrict__ Rwh) {
  const int idx = blockIdx.x * 256 + threadIdx.x;
  if (idx >= NB * CCH * HW) return;
  const int b = idx / (CCH * HW);
  const int c = (idx / HW) % CCH;
  const int h = (idx / WW) % HH;
  const int w = idx % WW;
  const int m = c >> 5, ci = c & 31;
  const unsigned char se = oe[idx] > 1.0f ? 1 : 0;
  const unsigned char sr = orr[idx] > 1.0f ? 1 : 0;
  const size_t hwI = ((size_t)(b * NHEADS + m) * HH + h) * DD + w * CHD + ci;
  const size_t whI = ((size_t)(b * NHEADS + m) * WW + w) * DD + h * CHD + ci;
  Ehw[hwI] = se; Ewh[whI] = se;
  Rhw[hwI] = sr; Rwh[whI] = sr;
}

__global__ __launch_bounds__(32)
void score_iu8_wmma_k(const unsigned char* __restrict__ Q,
                      const unsigned char* __restrict__ Km,
                      int* __restrict__ S) {
  const int bm = blockIdx.x, ti = blockIdx.y, tj = blockIdx.z;
  const int lane = threadIdx.x;
  const int lo = lane & 15, hi = lane >> 4;
  const unsigned char* qrow = Q + (size_t)(bm * 96 + ti * 16 + lo) * DD;
  const unsigned char* krow = Km + (size_t)(bm * 96 + tj * 16 + lo) * DD;
  v8i_t acc = {};
  for (int k0 = 0; k0 < DD; k0 += 64) {
    v8i_t a, b;
#pragma unroll
    for (int v = 0; v < 8; ++v) {
      const int ka = k0 + ((v >> 1) * 16 + hi * 8 + (v & 1) * 4);
      const int kb = k0 + ((v >> 1) * 16 + hi * 8 + (v & 1) * 4);
      a[v] = *(const int*)(qrow + ka);
      b[v] = *(const int*)(krow + kb);
    }
    acc = __builtin_amdgcn_wmma_i32_16x16x64_iu8(false, a, false, b, acc, false, false);
  }
#pragma unroll
  for (int r = 0; r < 8; ++r) {
    const int row = ti * 16 + r + hi * 8;
    const int col = tj * 16 + lo;
    S[(bm * 96 + row) * 96 + col] = acc[r];
  }
}

__global__ __launch_bounds__(128)
void softmax96_k(const int* __restrict__ S, float* __restrict__ P, int nrows) {
  const int row = blockIdx.x * 4 + (threadIdx.x >> 5);
  const int lane = threadIdx.x & 31;
  if (row >= nrows) return;
  const int* s = S + (size_t)row * 96;
  const float v0 = (float)s[lane], v1 = (float)s[lane + 32], v2 = (float)s[lane + 64];
  const float m = wred_max(fmaxf(v0, fmaxf(v1, v2)));
  const float e0 = __expf(v0 - m), e1 = __expf(v1 - m), e2 = __expf(v2 - m);
  const float inv = 1.0f / wred_sum(e0 + e1 + e2);
  float* p = P + (size_t)row * 96;
  p[lane] = e0 * inv; p[lane + 32] = e1 * inv; p[lane + 64] = e2 * inv;
}

__global__ __launch_bounds__(128)
void av_f16_wmma_k(const float* __restrict__ P, const unsigned char* __restrict__ V,
                   const unsigned char* __restrict__ Qa, float* __restrict__ O) {
  __shared__ unsigned char Vb[96 * 256];
  __shared__ _Float16 Pt[16 * 96];
  const int bm = blockIdx.x, ti = blockIdx.y, tjg = blockIdx.z;
  const int tid = threadIdx.x;
  const int c0 = tjg * 256;
#pragma unroll
  for (int i = 0; i < 12; ++i) {
    const int idx = tid + i * 128;
    const int r = idx >> 4;
    const int c16 = (idx & 15) << 4;
    const unsigned char* g = V + (size_t)(bm * 96 + r) * DD + c0 + c16;
    const unsigned lds = (unsigned)(size_t)(const void*)&Vb[r * 256 + c16];
    asm volatile("global_load_async_to_lds_b128 %0, %1, off"
                 :: "v"(lds), "v"(g) : "memory");
  }
#pragma unroll
  for (int i = 0; i < 12; ++i) {
    const int idx = tid + i * 128;
    const int r = idx / 96, cc = idx % 96;
    Pt[r * 96 + cc] = (_Float16)P[(size_t)(bm * 96 + ti * 16 + r) * 96 + cc];
  }
  asm volatile("s_wait_asynccnt 0x0" ::: "memory");
  __syncthreads();
  const int wid = tid >> 5, lane = tid & 31;
  const int lo = lane & 15, hi = lane >> 4;
  v8f_t acc[4] = {};
  for (int k0 = 0; k0 < 96; k0 += 32) {
    v16h_t a;
#pragma unroll
    for (int v = 0; v < 8; ++v) {
      const int ka = k0 + ((v >> 2) * 16 + hi * 8 + (v & 3) * 2);
      a[2 * v]     = Pt[lo * 96 + ka];
      a[2 * v + 1] = Pt[lo * 96 + ka + 1];
    }
#pragma unroll
    for (int q = 0; q < 4; ++q) {
      const int colL = (wid * 4 + q) * 16 + lo;
      v16h_t bmat;
#pragma unroll
      for (int v = 0; v < 8; ++v) {
        const int kb = k0 + ((v >> 2) * 16 + hi * 8 + (v & 3) * 2);
        bmat[2 * v]     = (_Float16)(float)Vb[kb * 256 + colL];
        bmat[2 * v + 1] = (_Float16)(float)Vb[(kb + 1) * 256 + colL];
      }
      acc[q] = __builtin_amdgcn_wmma_f32_16x16x32_f16(false, a, false, bmat, (short)0,
                                                      acc[q], false, false);
    }
  }
#pragma unroll
  for (int q = 0; q < 4; ++q) {
    const int col = c0 + (wid * 4 + q) * 16 + lo;
#pragma unroll
    for (int r = 0; r < 8; ++r) {
      const int row = ti * 16 + r + hi * 8;
      const size_t oi = (size_t)(bm * 96 + row) * DD + col;
      O[oi] = acc[q][r] + (float)Qa[oi];
    }
  }
}

__global__ __launch_bounds__(256)
void fuse_out_k(const float* __restrict__ O3, const float* __restrict__ O4,
                const float* __restrict__ xe, const float* __restrict__ xr,
                const float* __restrict__ pw, const float* __restrict__ pb,
                float* __restrict__ out_e, float* __restrict__ out_r,
                float* __restrict__ diff) {
  const int idx = blockIdx.x * 256 + threadIdx.x;
  if (idx >= NB * CCH * HW) return;
  const int b = idx / (CCH * HW);
  const int c = (idx / HW) % CCH;
  const int h = (idx / WW) % HH;
  const int w = idx % WW;
  const int m = c >> 5, ci = c & 31;
  const size_t hwI = ((size_t)(b * NHEADS + m) * HH + h) * DD + w * CHD + ci;
  const size_t whI = ((size_t)(b * NHEADS + m) * WW + w) * DD + h * CHD + ci;
  const float p3 = O3[hwI] * pw[c] + pb[c];
  const float p4 = O4[whI] * pw[c] + pb[c];
  const float common = p3 + 0.5f * p4;
  const float oe = common + xe[idx];
  const float orr = common + xr[idx];
  out_e[idx] = oe; out_r[idx] = orr; diff[idx] = oe - orr;
}

__global__ __launch_bounds__(256)
void dw3x3_k(const float* __restrict__ s0, const float* __restrict__ s1,
             const float* __restrict__ dww, const float* __restrict__ dwb,
             float* __restrict__ out) {
  const int idx = blockIdx.x * 256 + threadIdx.x;
  if (idx >= NB * 2 * CCH * HW) return;
  const int b  = idx / (2 * CCH * HW);
  const int cc = (idx / HW) % (2 * CCH);
  const int h  = (idx / WW) % HH;
  const int w  = idx % WW;
  const float* src = (cc < CCH) ? (s0 + ((size_t)b * CCH + cc) * HW)
                                : (s1 + ((size_t)b * CCH + (cc - CCH)) * HW);
  const float* wk = dww + cc * 9;
  float acc = dwb[cc];
#pragma unroll
  for (int dy = -1; dy <= 1; ++dy) {
    const int hh2 = h + dy;
    if ((unsigned)hh2 >= (unsigned)HH) continue;
#pragma unroll
    for (int dx = -1; dx <= 1; ++dx) {
      const int ww2 = w + dx;
      if ((unsigned)ww2 >= (unsigned)WW) continue;
      acc += src[hh2 * WW + ww2] * wk[(dy + 1) * 3 + (dx + 1)];
    }
  }
  out[idx] = acc;
}

#if HAVE_WMMA_F32
__global__ __launch_bounds__(128)
void pw_wmma_k(const float* __restrict__ in512, const float* __restrict__ pww,
               const float* __restrict__ pwb, const float* __restrict__ bng,
               const float* __restrict__ bnbb, const float* __restrict__ bnm,
               const float* __restrict__ bnv, float* __restrict__ out) {
  __shared__ float lin[512 * 16];
  const int b = blockIdx.x, mg = blockIdx.y, nt = blockIdx.z;
  const int pos0 = nt * 16;
  const int tid = threadIdx.x;
  {
    const int r0 = tid >> 2, c4 = (tid & 3) << 2;
#pragma unroll
    for (int it = 0; it < 16; ++it) {
      const int ci = r0 + it * 32;
      const float4 vv = *(const float4*)(in512 + ((size_t)b * 2 * CCH + ci) * HW + pos0 + c4);
      lin[ci * 16 + c4 + 0] = vv.x; lin[ci * 16 + c4 + 1] = vv.y;
      lin[ci * 16 + c4 + 2] = vv.z; lin[ci * 16 + c4 + 3] = vv.w;
    }
  }
  __builtin_prefetch(in512 + ((size_t)b * 2 * CCH) * HW + pos0 + 16, 0, 0);
  __syncthreads();
  const int wid = tid >> 5, lane = tid & 31;
  const int lo = lane & 15, hi = lane >> 4;
  const int m0 = mg * 64 + wid * 16;
  const float* wrow = pww + (size_t)(m0 + lo) * 2 * CCH;
  v8f_t acc0 = {}, acc1 = {};
  for (int k0 = 0; k0 < 2 * CCH; k0 += 8) {
    v2f_t aa0, bb0, aa1, bb1;
    const int ka = k0 + 2 * hi;
    aa0[0] = wrow[ka];               aa0[1] = wrow[ka + 1];
    bb0[0] = lin[ka * 16 + lo];      bb0[1] = lin[(ka + 1) * 16 + lo];
    const int kc = k0 + 4 + 2 * hi;
    aa1[0] = wrow[kc];               aa1[1] = wrow[kc + 1];
    bb1[0] = lin[kc * 16 + lo];      bb1[1] = lin[(kc + 1) * 16 + lo];
    acc0 = __builtin_amdgcn_wmma_f32_16x16x4_f32(false, aa0, false, bb0, (short)0, acc0,
                                                 false, false);
    acc1 = __builtin_amdgcn_wmma_f32_16x16x4_f32(false, aa1, false, bb1, (short)0, acc1,
                                                 false, false);
  }
#pragma unroll
  for (int r = 0; r < 8; ++r) {
    const int co = m0 + r + hi * 8;
    const float a = acc0[r] + acc1[r] + pwb[co];
    const float y = (a - bnm[co]) * rsqrtf(bnv[co] + 1e-5f) * bng[co] + bnbb[co];
    out[((size_t)b * CCH + co) * HW + pos0 + lo] = fmaxf(y, 0.f);
  }
}
#endif

__global__ __launch_bounds__(256)
void pw_bn_relu_k(const float* __restrict__ in512, const float* __restrict__ pww,
                  const float* __restrict__ pwb, const float* __restrict__ bng,
                  const float* __restrict__ bnbb, const float* __restrict__ bnm,
                  const float* __restrict__ bnv, float* __restrict__ out) {
  const int idx = blockIdx.x * 256 + threadIdx.x;
  if (idx >= NB * CCH * HW) return;
  const int b   = idx / (CCH * HW);
  const int co  = (idx / HW) % CCH;
  const int pos = idx % HW;
  const float* in = in512 + (size_t)b * 2 * CCH * HW + pos;
  const float* wk = pww + co * 2 * CCH;
  float acc = pwb[co];
  for (int ci = 0; ci < 2 * CCH; ++ci) acc += in[(size_t)ci * HW] * wk[ci];
  const float y = (acc - bnm[co]) * rsqrtf(bnv[co] + 1e-5f) * bng[co] + bnbb[co];
  out[idx] = fmaxf(y, 0.f);
}

__global__ __launch_bounds__(256)
void chanpool_sigmoid_k(const float* __restrict__ x, const float* __restrict__ wk,
                        const float* __restrict__ b0, float* __restrict__ map) {
  const int idx = blockIdx.x * 256 + threadIdx.x;
  if (idx >= NB * HW) return;
  const int b = idx / HW, pos = idx % HW;
  const float* in = x + (size_t)b * CCH * HW + pos;
  float acc = b0[0];
  for (int c = 0; c < CCH; ++c) acc += in[(size_t)c * HW] * wk[c];
  map[idx] = 1.0f / (1.0f + __expf(-acc));
}

__global__ __launch_bounds__(256)
void gate_scale_k(const float* __restrict__ map, const float* __restrict__ src,
                  float* __restrict__ out) {
  const int idx = blockIdx.x * 256 + threadIdx.x;
  if (idx >= NB * CCH * HW) return;
  const int b = idx / (CCH * HW);
  const int pos = idx % HW;
  out[idx] = map[b * HW + pos] * src[idx];
}

__global__ __launch_bounds__(256)
void final_lif_k(const float* __restrict__ x, float* __restrict__ out) {
  const int idx = blockIdx.x * 256 + threadIdx.x;
  if (idx >= NB * CCH * HW) return;
  const float v = x[idx];
  out[idx] = (v > 1.0f) ? v + 1.0f : v;
}

extern "C" void kernel_launch(void* const* d_in, const int* in_sizes, int n_in,
                              void* d_out, int out_size, void* d_ws, size_t ws_size,
                              hipStream_t stream) {
  (void)in_sizes; (void)n_in; (void)out_size; (void)ws_size;
  const float* ev   = (const float*)d_in[0];
  const float* rgb  = (const float*)d_in[1];
  const float* ln_w = (const float*)d_in[2];
  const float* ln_b = (const float*)d_in[3];
  const float* pjw  = (const float*)d_in[4];
  const float* pjb  = (const float*)d_in[5];
  const float* fc1  = (const float*)d_in[6];
  const float* fc2  = (const float*)d_in[7];
  const float* cAw  = (const float*)d_in[8];
  const float* cAb  = (const float*)d_in[9];
  const float* cBw  = (const float*)d_in[10];
  const float* cBb  = (const float*)d_in[11];
  const float* rW[6], *rB[6];
  for (int i = 0; i < 6; ++i) { rW[i] = (const float*)d_in[12 + 2 * i]; rB[i] = (const float*)d_in[13 + 2 * i]; }
  const float* cat[3][8];
  for (int t = 0; t < 3; ++t)
    for (int j = 0; j < 8; ++j) cat[t][j] = (const float*)d_in[24 + t * 8 + j];

  const size_t TE = (size_t)NB * CCH * HW;
  const size_t T  = TE * sizeof(float);
  const size_t U  = (size_t)NB * NHEADS * 96 * DD;
  const size_t SB = (size_t)NB * NHEADS * 96 * 96 * sizeof(int);
  char* ws = (char*)d_ws;

  float* p_xe   = (float*)(ws);
  float* p_xr   = (float*)(ws + T);
  float* p_oute = (float*)(ws + 2 * T);
  float* p_outr = (float*)(ws + 3 * T);
  unsigned char* p_Ehw = (unsigned char*)(ws + 4 * T);
  unsigned char* p_Rhw = p_Ehw + U;
  unsigned char* p_Ewh = p_Rhw + U;
  unsigned char* p_Rwh = p_Ewh + U;
  char* p_small = ws + 4 * T + 4 * U;
  float* p_gap  = (float*)(p_small);
  float* p_att  = (float*)(p_small + 4096);
  float* p_mapA = (float*)(p_small + 8192);
  float* p_mapB = (float*)(p_small + 8192 + NB * HW * sizeof(float));
  char*  p_big  = p_small + (1 << 20);
  int*   p_S    = (int*)(p_big);
  float* p_P    = (float*)(p_big + SB);
  float* p_O3   = (float*)(p_big + 2 * SB);
  float* p_O4   = (float*)(p_big + 2 * SB + T);
  float* p_diff = (float*)p_Ehw;
  float* p_512  = p_O3;
  float* p_xA   = p_xe;
  float* p_xB   = p_xr;
  float* p_dsA  = p_oute;
  float* p_dsB  = p_outr;

  float* out_event = (float*)d_out;
  float* out_rgb   = out_event + TE;
  float* out_fus   = out_event + 2 * TE;

  const int EB  = (int)((TE + 255) / 256);
  const int EB2 = (int)((2 * TE + 255) / 256);

  layernorm_k<<<NB * HW, 256, 0, stream>>>(ev, ln_w, ln_b, p_xe);
  layernorm_k<<<NB * HW, 256, 0, stream>>>(rgb, ln_w, ln_b, p_xr);

  gap_k<<<NB * CCH, 256, 0, stream>>>(p_xe, p_gap);
  se_fc_k<<<NB, 256, 0, stream>>>(p_gap, fc1, fc2, p_att);
  oute_k<<<EB, 256, 0, stream>>>(p_xe, p_att, pjw, pjb, p_oute);

  outr_k<<<EB, 256, 0, stream>>>(p_xr, rW[0], rB[0], rW[1], rB[1], rW[2], rB[2],
                                 rW[3], rB[3], rW[4], rB[4], rW[5], rB[5],
                                 pjw, pjb, p_outr);

  spike_rearrange_k<<<EB, 256, 0, stream>>>(p_oute, p_outr, p_Ehw, p_Rhw, p_Ewh, p_Rwh);

  {
    dim3 gs(NB * NHEADS, 6, 6), ga(NB * NHEADS, 6, 12);
    score_iu8_wmma_k<<<gs, 32, 0, stream>>>(p_Rhw, p_Ehw, p_S);
    softmax96_k<<<(NB * NHEADS * 96 + 3) / 4, 128, 0, stream>>>(p_S, p_P, NB * NHEADS * 96);
    av_f16_wmma_k<<<ga, 128, 0, stream>>>(p_P, p_Ehw, p_Rhw, p_O3);
  }
  {
    dim3 gs(NB * NHEADS, 6, 6), ga(NB * NHEADS, 6, 12);
    score_iu8_wmma_k<<<gs, 32, 0, stream>>>(p_Ewh, p_Rwh, p_S);
    softmax96_k<<<(NB * NHEADS * 96 + 3) / 4, 128, 0, stream>>>(p_S, p_P, NB * NHEADS * 96);
    av_f16_wmma_k<<<ga, 128, 0, stream>>>(p_P, p_Rwh, p_Ewh, p_O4);
  }

  fuse_out_k<<<EB, 256, 0, stream>>>(p_O3, p_O4, p_xe, p_xr, pjw, pjb,
                                     out_event, out_rgb, p_diff);

#if HAVE_WMMA_F32
  #define PW_LAUNCH(t, in, outp) \
    pw_wmma_k<<<dim3(NB, 4, 576), 128, 0, stream>>>(in, cat[t][2], cat[t][3], \
        cat[t][4], cat[t][5], cat[t][6], cat[t][7], outp)
#else
  #define PW_LAUNCH(t, in, outp) \
    pw_bn_relu_k<<<EB, 256, 0, stream>>>(in, cat[t][2], cat[t][3], \
        cat[t][4], cat[t][5], cat[t][6], cat[t][7], outp)
#endif

  dw3x3_k<<<EB2, 256, 0, stream>>>(p_diff, out_event, cat[0][0], cat[0][1], p_512);
  PW_LAUNCH(0, p_512, p_dsA);
  chanpool_sigmoid_k<<<(NB * HW + 255) / 256, 256, 0, stream>>>(p_dsA, cAw, cAb, p_mapA);
  gate_scale_k<<<EB, 256, 0, stream>>>(p_mapA, out_event, p_xA);

  dw3x3_k<<<EB2, 256, 0, stream>>>(p_diff, out_rgb, cat[1][0], cat[1][1], p_512);
  PW_LAUNCH(1, p_512, p_dsB);
  chanpool_sigmoid_k<<<(NB * HW + 255) / 256, 256, 0, stream>>>(p_dsB, cBw, cBb, p_mapB);
  gate_scale_k<<<EB, 256, 0, stream>>>(p_mapB, out_rgb, p_xB);

  dw3x3_k<<<EB2, 256, 0, stream>>>(p_xA, p_xB, cat[2][0], cat[2][1], p_512);
  PW_LAUNCH(2, p_512, p_dsA);
  final_lif_k<<<EB, 256, 0, stream>>>(p_dsA, out_fus);
}
